// GCNEncoder_8108898255681
// MI455X (gfx1250) — compile-verified, matrix operand order checked by reading, not run on hardware
//
#include <hip/hip_runtime.h>
#include <hip/hip_bf16.h>

typedef __attribute__((ext_vector_type(16))) _Float16 v16h;
typedef __attribute__((ext_vector_type(8)))  float    v8f;

// ---------------- degree / norm kernels ----------------

__global__ void k_fill1(float* __restrict__ deg, int n) {
    int i = blockIdx.x * blockDim.x + threadIdx.x;
    if (i < n) deg[i] = 1.0f;   // self-loop contributes 1
}

__global__ void k_degree(const int* __restrict__ dst, float* __restrict__ deg, int E) {
    int e = blockIdx.x * blockDim.x + threadIdx.x;
    if (e < E) atomicAdd(&deg[dst[e]], 1.0f);
}

__global__ void k_rsqrt_inplace(float* __restrict__ deg, int n) {
    int i = blockIdx.x * blockDim.x + threadIdx.x;
    if (i < n) deg[i] = rsqrtf(deg[i]);   // deg buffer becomes dinv
}

// ---------------- WMMA GEMM: H[N x OUTC] = act(X[N x 128]) * W[128 x OUTC] ----------------
// Reads FP32 operands directly and converts to FP16 in-register (pairs are
// contiguous, so A-fragment loads coalesce into b64/b128 vector loads).
// W is converted + transposed into LDS as [n][k] so B-fragment K-pairs are
// contiguous ds_load_b32/b128. One wave computes one 16-row tile across OUTC.

template <int OUTC, bool RELU>
__global__ __launch_bounds__(256)
void k_gemm_f16(const float* __restrict__ X, const float* __restrict__ Wg,
                float* __restrict__ H, int ntiles) {
    __shared__ _Float16 ldsW[OUTC * 128];   // [n][k]; 32KB (OUTC=128) / 16KB (OUTC=64)

    const int tid = threadIdx.x;
    for (int idx = tid; idx < 128 * OUTC; idx += 256) {
        int k = idx / OUTC;
        int n = idx % OUTC;
        ldsW[n * 128 + k] = (_Float16)Wg[idx];     // fp32 -> fp16 + transpose
    }
    __syncthreads();

    const int wave = tid >> 5;
    const int lane = tid & 31;
    const int tile = blockIdx.x * 8 + wave;
    if (tile >= ntiles) return;             // wave-uniform: EXEC stays all-1s

    const int mn = lane & 15;               // M for A-frag, N for B/C-frags
    const int hh = lane >> 4;               // lane half selects K(+8) / M(+8)

    // A fragments: 16x32 f16 per K-step, 4 steps cover K=128.
    // VGPR i holds K-pair (2i or 2i+8) + 8*hh within the 32-wide K block.
    const float* xrow = X + (size_t)(tile * 16 + mn) * 128;
    v16h a[4];
#pragma unroll
    for (int kf = 0; kf < 4; ++kf) {
        const int k0 = kf * 32;
#pragma unroll
        for (int i = 0; i < 8; ++i) {
            int kk = k0 + ((i < 4) ? 2 * i : 2 * i + 8) + 8 * hh;
            float v0 = xrow[kk];
            float v1 = xrow[kk + 1];
            if (RELU) { v0 = fmaxf(v0, 0.0f); v1 = fmaxf(v1, 0.0f); }
            a[kf][2 * i]     = (_Float16)v0;
            a[kf][2 * i + 1] = (_Float16)v1;
        }
    }

#pragma unroll
    for (int nt = 0; nt < OUTC / 16; ++nt) {
        const int col = nt * 16 + mn;
        const _Float16* wrow = &ldsW[col * 128];
        v8f c = {};
#pragma unroll
        for (int kf = 0; kf < 4; ++kf) {
            const int k0 = kf * 32;
            v16h b;
#pragma unroll
            for (int i = 0; i < 8; ++i) {
                int kk = k0 + ((i < 4) ? 2 * i : 2 * i + 8) + 8 * hh;
                b[2 * i]     = wrow[kk];
                b[2 * i + 1] = wrow[kk + 1];
            }
            c = __builtin_amdgcn_wmma_f32_16x16x32_f16(
                    false, a[kf], false, b, (short)0, c, false, false);
        }
        // C/D layout: VGPR r -> row (r + 8*hh), lane 0..15 -> col
        float* hout = H + (size_t)(tile * 16) * OUTC + col;
#pragma unroll
        for (int r = 0; r < 8; ++r)
            hout[(size_t)(r + 8 * hh) * OUTC] = c[r];
    }
}

// ---------------- self-loop + bias init: agg = h * dinv^2 + b ----------------

template <int C>
__global__ void k_selfbias(const float* __restrict__ h, const float* __restrict__ dinv,
                           const float* __restrict__ bias, float* __restrict__ agg, int n) {
    constexpr int G = C / 4;
    int t = blockIdx.x * blockDim.x + threadIdx.x;
    int i = t / G;
    int g = t % G;
    if (i >= n) return;
    float di = dinv[i];
    float s  = di * di;
    const float4 hv = *(const float4*)(h + (size_t)i * C + g * 4);
    const float4 bv = *(const float4*)(bias + g * 4);
    float4 o;
    o.x = hv.x * s + bv.x;
    o.y = hv.y * s + bv.y;
    o.z = hv.z * s + bv.z;
    o.w = hv.w * s + bv.w;
    *(float4*)(agg + (size_t)i * C + g * 4) = o;
}

// ---------------- edge gather-scale-scatter (bandwidth-bound part) ----------------
// one thread per (edge, 4-channel group): a wave covers one edge's full row
// (coalesced float4 gather + coalesced f32 global atomics)

template <int C>
__global__ void k_scatter(const float* __restrict__ h, const int* __restrict__ src,
                          const int* __restrict__ dst, const float* __restrict__ dinv,
                          float* __restrict__ agg, long long total) {
    constexpr int G = C / 4;
    long long t = (long long)blockIdx.x * blockDim.x + threadIdx.x;
    if (t >= total) return;
    int e = (int)(t / G);
    int g = (int)(t % G);
    int s = src[e];
    int d = dst[e];
    float nrm = dinv[s] * dinv[d];
    const float4 hv = *(const float4*)(h + (size_t)s * C + g * 4);
    float* p = agg + (size_t)d * C + g * 4;
    atomicAdd(p + 0, hv.x * nrm);
    atomicAdd(p + 1, hv.y * nrm);
    atomicAdd(p + 2, hv.z * nrm);
    atomicAdd(p + 3, hv.w * nrm);
}

// ---------------- launch ----------------

extern "C" void kernel_launch(void* const* d_in, const int* in_sizes, int n_in,
                              void* d_out, int out_size, void* d_ws, size_t ws_size,
                              hipStream_t stream) {
    const float* x  = (const float*)d_in[0];
    const int*   ei = (const int*)d_in[1];
    const float* W1 = (const float*)d_in[2];
    const float* b1 = (const float*)d_in[3];
    const float* W2 = (const float*)d_in[4];
    const float* b2 = (const float*)d_in[5];
    float*       out = (float*)d_out;

    const int N = in_sizes[0] / 128;   // 100000
    const int E = in_sizes[1] / 2;     // 1600000
    const int* src = ei;
    const int* dst = ei + E;

    // carve workspace (256B aligned); reuse dead buffers
    char* w = (char*)d_ws;
    auto carve = [&](size_t bytes) -> void* {
        void* p = (void*)w;
        w += (bytes + 255) & ~(size_t)255;
        return p;
    };
    float* dinv = (float*)carve((size_t)N * 4);          // deg -> dinv in place
    float* h1   = (float*)carve((size_t)N * 128 * 4);
    float* agg1 = (float*)carve((size_t)N * 128 * 4);
    float* h2   = h1;   // h1 dead after layer-1 scatter

    const int B = 256;
    const int ntiles = (N + 15) / 16;                 // 6250
    const int gemm_blocks = (ntiles + 7) / 8;

    // degrees / symmetric norm
    k_fill1<<<(N + B - 1) / B, B, 0, stream>>>(dinv, N);
    k_degree<<<(E + B - 1) / B, B, 0, stream>>>(dst, dinv, E);
    k_rsqrt_inplace<<<(N + B - 1) / B, B, 0, stream>>>(dinv, N);

    // layer 1: h1 = x @ W1 ; agg1 = scatter(norm*h1) + h1*dinv^2 + b1
    k_gemm_f16<128, false><<<gemm_blocks, 256, 0, stream>>>(x, W1, h1, ntiles);
    k_selfbias<128><<<(N * 32 + B - 1) / B, B, 0, stream>>>(h1, dinv, b1, agg1, N);
    {
        long long total = (long long)E * 32;
        k_scatter<128><<<(int)((total + B - 1) / B), B, 0, stream>>>(h1, src, dst, dinv, agg1, total);
    }

    // layer 2: h2 = relu(agg1) @ W2 ; out = scatter(norm*h2) + h2*dinv^2 + b2
    k_gemm_f16<64, true><<<gemm_blocks, 256, 0, stream>>>(agg1, W2, h2, ntiles);
    k_selfbias<64><<<(N * 16 + B - 1) / B, B, 0, stream>>>(h2, dinv, b2, out, N);
    {
        long long total = (long long)E * 16;
        k_scatter<64><<<(int)((total + B - 1) / B), B, 0, stream>>>(h2, src, dst, dinv, out, total);
    }
}
